// Stage2CandidateGraphSelector_15333033246793
// MI455X (gfx1250) — matrix operand order repaired and checked by reading, not compiled, not run on hardware
//
#include <hip/hip_runtime.h>
#include <hip/hip_bf16.h>

typedef __attribute__((ext_vector_type(16))) _Float16 v16h;
typedef __attribute__((ext_vector_type(8)))  _Float16 v8h;
typedef __attribute__((ext_vector_type(8)))  float    v8f;

#define WMMA_F16(a, b, c) \
  __builtin_amdgcn_wmma_f32_16x16x32_f16(false, (a), false, (b), (short)0, (c), false, false)

// ---- WMMA fragment helpers (CDNA5 ISA 7.12.2 layouts, wave32) ----
// A 16x32 f16: lane holds row M=(lane&15); element i -> K = i + kb (i<8) or
// i+8+kb (i>=8), kb=(lane&16)?8:0 -> two contiguous 8-half chunks at
// [kb, kb+8) and [16+kb, 16+kb+8).

__device__ __forceinline__ v16h combine16(v8h lo, v8h hi) {
  union { v16h v; v8h h[2]; } u;
  u.h[0] = lo; u.h[1] = hi;
  return u.v;
}

// A fragment from a row-major f16 matrix in LDS (lda halves, multiple of 8).
__device__ __forceinline__ v16h load_a_lds(const _Float16* base, int lda, int kstep, int lane) {
  const _Float16* p = base + (size_t)(lane & 15) * lda + kstep * 32 + ((lane & 16) ? 8 : 0);
  return combine16(*(const v8h*)p, *(const v8h*)(p + 16));
}

// B fragment from fragment-order packed weights: 16 contiguous halves per lane.
__device__ __forceinline__ v16h load_b_pk(const _Float16* Wpk, int kstep, int nt, int lane) {
  const _Float16* p = Wpk + ((((size_t)kstep * 4 + nt) * 32) + lane) * 16;
  return combine16(*(const v8h*)p, *(const v8h*)(p + 8));
}

// C/D 16x16 f32: VGPR r -> row (mbase + r + (lane&16?8:0)), col = nt*16+(lane&15).
__device__ __forceinline__ void store_c_relu_f16(_Float16* dst, int ldd, v8f c,
                                                 const float* bias, int mbase, int nt, int lane) {
  int n  = nt * 16 + (lane & 15);
  float bn = bias[n];
  int rb = mbase + ((lane & 16) ? 8 : 0);
#pragma unroll
  for (int r = 0; r < 8; ++r)
    dst[(size_t)(rb + r) * ldd + n] = (_Float16)fmaxf(c[r] + bn, 0.f);
}

// ---- Kernel 0: convert + pack weights into WMMA B-fragment order ----
// packed[((ks*4+nt)*32+lane)*16 + i] = W[ks*32 + (lane&16?8:0) + i + (i>=8?8:0)][nt*16 + (lane&15)]
__device__ __forceinline__ void pack_b(_Float16* dst, const float* src, int realK, int idx) {
  int i    = idx & 15;
  int frag = idx >> 4;
  int lane = frag & 31;
  int ntks = frag >> 5;
  int nt = ntks & 3, ks = ntks >> 2;
  int k = ks * 32 + ((lane & 16) ? 8 : 0) + i + ((i >= 8) ? 8 : 0);
  int n = nt * 16 + (lane & 15);
  dst[idx] = (_Float16)((k < realK) ? src[k * 64 + n] : 0.f);
}

__global__ void prep_weights(const float* lp_w1, const float* lp_w2,
                             const float* np_w1, const float* np_w2,
                             const float* ep_w1, const float* ep_w2,
                             const float* sc_w1,
                             _Float16* lpW1p, _Float16* lpW2p,
                             _Float16* npW1p, _Float16* npW2p,
                             _Float16* epW1p, _Float16* epW2p,
                             _Float16* scW1p) {
  int idx = blockIdx.x * blockDim.x + threadIdx.x;
  const int N0 = 512 * 64, N1 = 64 * 64, N2 = 96 * 64, N3 = 64 * 64,
            N4 = 160 * 64, N5 = 64 * 64, N6 = 128 * 64;
  if (idx < N0) { pack_b(lpW1p, lp_w1, 512, idx); return; }
  idx -= N0;
  if (idx < N1) { pack_b(lpW2p, lp_w2, 64, idx); return; }
  idx -= N1;
  if (idx < N2) { pack_b(npW1p, np_w1, 80, idx); return; }
  idx -= N2;
  if (idx < N3) { pack_b(npW2p, np_w2, 64, idx); return; }
  idx -= N3;
  if (idx < N4) { pack_b(epW1p, ep_w1, 132, idx); return; }
  idx -= N4;
  if (idx < N5) { pack_b(epW2p, ep_w2, 64, idx); return; }
  idx -= N5;
  if (idx < N6) { pack_b(scW1p, sc_w1, 128, idx); }
}

// ---- Kernel 1: lang mean-pool (streaming, memory-bound) + 512->64->64 MLP ----
__global__ void lang_kernel(const float* __restrict__ lang,
                            const float* lp_b1, const float* lp_b2,
                            const _Float16* lpW1p, const _Float16* lpW2p,
                            _Float16* langH) {
  __shared__ _Float16 sMean[32][512];   // 32 KB
  __shared__ _Float16 sH1[32][64];      // 4 KB
  int tid = threadIdx.x, lane = tid & 31, wave = tid >> 5;
  int brow = blockIdx.x * 32;

  for (int p = tid; p < 32 * 512; p += 256) {
    int r = p >> 9, d = p & 511;
    const float* src = lang + ((size_t)(brow + r) * 77) * 512 + d;
    float acc = 0.f;
    for (int l = 0; l < 77; ++l) acc += src[(size_t)l * 512];
    sMean[r][d] = (_Float16)(acc * (1.0f / 77.0f));
  }
  __syncthreads();

  int mt = wave >> 2, nt = wave & 3;   // 8 waves: 2 M-tiles x 4 N-tiles
  { // layer 1: K = 512
    v8f c = {};
#pragma unroll 4
    for (int ks = 0; ks < 16; ++ks) {
      v16h a = load_a_lds(&sMean[mt * 16][0], 512, ks, lane);
      v16h b = load_b_pk(lpW1p, ks, nt, lane);
      c = WMMA_F16(a, b, c);
    }
    store_c_relu_f16(&sH1[0][0], 64, c, lp_b1, mt * 16, nt, lane);
  }
  __syncthreads();
  { // layer 2: K = 64
    v8f c = {};
#pragma unroll
    for (int ks = 0; ks < 2; ++ks) {
      v16h a = load_a_lds(&sH1[mt * 16][0], 64, ks, lane);
      v16h b = load_b_pk(lpW2p, ks, nt, lane);
      c = WMMA_F16(a, b, c);
    }
    store_c_relu_f16(langH + (size_t)brow * 64, 64, c, lp_b2, mt * 16, nt, lane);
  }
}

// ---- Kernel 2: node feature assembly + 96(pad of 80)->64->64 MLP ----
__global__ void node_kernel(const float* xyz, const float* scores, const float* proprio,
                            const float* view, const _Float16* langH,
                            const float* np_b1, const float* np_b2,
                            const _Float16* npW1p, const _Float16* npW2p,
                            _Float16* nodeH) {
  __shared__ _Float16 sIn[32][96];
  __shared__ _Float16 sH1[32][64];
  int tid = threadIdx.x, lane = tid & 31, wave = tid >> 5;
  int rowbase = blockIdx.x * 32;   // rows over B*K

  // scalar features (d < 16) + pad (d >= 80)
  for (int p = tid; p < 32 * 32; p += 256) {
    int r = p >> 5, d = p & 31;
    int gr = rowbase + r;
    int b = gr / 3, k = gr % 3;
    float v;
    if (d < 3)        v = xyz[gr * 3 + d];
    else if (d == 3)  v = scores[gr];
    else if (d < 7)   v = (d - 4 == k) ? 1.f : 0.f;        // rank one-hot
    else if (d < 12)  v = view[gr * 5 + (d - 7)];
    else if (d < 16)  v = proprio[b * 4 + (d - 12)];
    else              v = 0.f;
    sIn[r][(d < 16) ? d : (d + 64)] = (_Float16)v;         // d16..31 -> pad cols 80..95
  }
  // lang_h block (cols 16..79), vectorized 8-half copies
  for (int p = tid; p < 32 * 8; p += 256) {
    int r = p >> 3, c8 = p & 7;
    int b = (rowbase + r) / 3;
    *(v8h*)&sIn[r][16 + c8 * 8] = *(const v8h*)&langH[(size_t)b * 64 + c8 * 8];
  }
  __syncthreads();

  int mt = wave >> 2, nt = wave & 3;
  { v8f c = {};
#pragma unroll
    for (int ks = 0; ks < 3; ++ks) {
      v16h a = load_a_lds(&sIn[mt * 16][0], 96, ks, lane);
      v16h b = load_b_pk(npW1p, ks, nt, lane);
      c = WMMA_F16(a, b, c);
    }
    store_c_relu_f16(&sH1[0][0], 64, c, np_b1, mt * 16, nt, lane);
  }
  __syncthreads();
  { v8f c = {};
#pragma unroll
    for (int ks = 0; ks < 2; ++ks) {
      v16h a = load_a_lds(&sH1[mt * 16][0], 64, ks, lane);
      v16h b = load_b_pk(npW2p, ks, nt, lane);
      c = WMMA_F16(a, b, c);
    }
    store_c_relu_f16(nodeH + (size_t)rowbase * 64, 64, c, np_b2, mt * 16, nt, lane);
  }
}

// ---- Kernel 3: edge MLP 160(pad of 132)->64->64 + masked mean pooling ----
// 16 batch elems per block -> 144 edge rows = 9 M-tiles; 4 N-tiles; 8 waves loop 36 tasks.
__global__ void edge_kernel(const float* xyz, const _Float16* nodeH,
                            const float* ep_b1, const float* ep_b2,
                            const _Float16* epW1p, const _Float16* epW2p,
                            _Float16* pooledH) {
  __shared__ _Float16 sNode[48][64];  // 6 KB  : this block's node rows
  __shared__ _Float16 sTail[144][32]; // 9 KB  : [dx,dy,dz,dist,0...] per edge row
  __shared__ _Float16 sH1[144][64];   // 18 KB
  __shared__ float    sPool[48][64];  // 12 KB
  int tid = threadIdx.x, lane = tid & 31, wave = tid >> 5;
  int bbase = blockIdx.x * 16;

  // stage node rows (vectorized) and clear pool
  for (int p = tid; p < 48 * 8; p += 256) {
    int r = p >> 3, c8 = p & 7;
    *(v8h*)&sNode[r][c8 * 8] = *(const v8h*)&nodeH[((size_t)bbase * 3 + r) * 64 + c8 * 8];
  }
  for (int p = tid; p < 48 * 64; p += 256) sPool[p >> 6][p & 63] = 0.f;
  // geometry tail rows
  for (int lr = tid; lr < 144; lr += 256) {
    int b   = bbase + lr / 9;
    int rem = lr % 9;
    int ei = rem / 3, ej = rem % 3;
    float dx = xyz[(b * 3 + ei) * 3 + 0] - xyz[(b * 3 + ej) * 3 + 0];
    float dy = xyz[(b * 3 + ei) * 3 + 1] - xyz[(b * 3 + ej) * 3 + 1];
    float dz = xyz[(b * 3 + ei) * 3 + 2] - xyz[(b * 3 + ej) * 3 + 2];
    float dist = sqrtf(dx * dx + dy * dy + dz * dz);
    v8h z = {};
    *(v8h*)&sTail[lr][8]  = z;
    *(v8h*)&sTail[lr][16] = z;
    *(v8h*)&sTail[lr][24] = z;
    sTail[lr][0] = (_Float16)dx; sTail[lr][1] = (_Float16)dy;
    sTail[lr][2] = (_Float16)dz; sTail[lr][3] = (_Float16)dist;
    sTail[lr][4] = (_Float16)0.f; sTail[lr][5] = (_Float16)0.f;
    sTail[lr][6] = (_Float16)0.f; sTail[lr][7] = (_Float16)0.f;
  }
  __syncthreads();

  // Layer 1: A chunks resolved to one of three LDS regions, all 16B vector loads.
  for (int t = wave; t < 36; t += 8) {
    int mt = t >> 2, nt = t & 3;
    int lr = mt * 16 + (lane & 15);      // this lane's A row, 0..143
    int lb  = lr / 9;
    int rem = lr % 9;
    int ei = rem / 3, ej = rem % 3;
    const _Float16* ni   = &sNode[lb * 3 + ei][0];
    const _Float16* nj   = &sNode[lb * 3 + ej][0];
    const _Float16* tail = &sTail[lr][0];
    int kb = (lane & 16) ? 8 : 0;
    v8f c = {};
#pragma unroll
    for (int ks = 0; ks < 5; ++ks) {
      int o0 = ks * 32 + kb, o1 = o0 + 16;
      const _Float16* p0 = (o0 < 64) ? (ni + o0) : (o0 < 128) ? (nj + o0 - 64) : (tail + o0 - 128);
      const _Float16* p1 = (o1 < 64) ? (ni + o1) : (o1 < 128) ? (nj + o1 - 64) : (tail + o1 - 128);
      v16h a  = combine16(*(const v8h*)p0, *(const v8h*)p1);
      v16h bf = load_b_pk(epW1p, ks, nt, lane);
      c = WMMA_F16(a, bf, c);
    }
    store_c_relu_f16(&sH1[0][0], 64, c, ep_b1, mt * 16, nt, lane);
  }
  __syncthreads();

  // Layer 2 + masked pooling (skip self-edges) via LDS float atomics.
  for (int t = wave; t < 36; t += 8) {
    int mt = t >> 2, nt = t & 3;
    v8f c = {};
#pragma unroll
    for (int ks = 0; ks < 2; ++ks) {
      v16h a  = load_a_lds(&sH1[mt * 16][0], 64, ks, lane);
      v16h bf = load_b_pk(epW2p, ks, nt, lane);
      c = WMMA_F16(a, bf, c);
    }
    int n  = nt * 16 + (lane & 15);
    float bn = ep_b2[n];
    int rb = mt * 16 + ((lane & 16) ? 8 : 0);
#pragma unroll
    for (int r = 0; r < 8; ++r) {
      int row = rb + r;
      int lb = row / 9, rem = row % 9;
      int ei = rem / 3, ej = rem % 3;
      if (ei != ej) {
        float v = fmaxf(c[r] + bn, 0.f);
        atomicAdd(&sPool[lb * 3 + ei][n], v);   // ds_add_f32
      }
    }
  }
  __syncthreads();

  for (int p = tid; p < 48 * 64; p += 256) {
    int r = p >> 6, n = p & 63;
    pooledH[((size_t)bbase * 3 + r) * 64 + n] = (_Float16)(sPool[r][n] * 0.5f); // / (K-1)
  }
}

// ---- Kernel 4: scorer 128->64 (WMMA) with fused 64->1 projection ----
__global__ void score_kernel(const _Float16* nodeH, const _Float16* pooledH,
                             const float* sc_b1, const float* sc_w2, const float* sc_b2,
                             const _Float16* scW1p, float* out) {
  __shared__ _Float16 sA[64][128];   // 16 KB: [node | pooled]
  int tid = threadIdx.x, lane = tid & 31, wave = tid >> 5;
  int blockrow = blockIdx.x * 64;

  for (int p = tid; p < 64 * 16; p += 128) {
    int r = p >> 4, c8 = p & 15;
    const _Float16* src = (c8 < 8)
        ? &nodeH[((size_t)blockrow + r) * 64 + c8 * 8]
        : &pooledH[((size_t)blockrow + r) * 64 + (c8 - 8) * 8];
    *(v8h*)&sA[r][c8 * 8] = *(const v8h*)src;
  }
  __syncthreads();

  int rowbase = wave * 16;     // each wave owns a 16-row tile of the 64
  float dacc[8];
#pragma unroll
  for (int r = 0; r < 8; ++r) dacc[r] = 0.f;

  for (int nt = 0; nt < 4; ++nt) {
    v8f c = {};
#pragma unroll
    for (int ks = 0; ks < 4; ++ks) {
      v16h a = load_a_lds(&sA[rowbase][0], 128, ks, lane);
      v16h b = load_b_pk(scW1p, ks, nt, lane);
      c = WMMA_F16(a, b, c);
    }
    int n = nt * 16 + (lane & 15);
    float bn = sc_b1[n], wn = sc_w2[n];
#pragma unroll
    for (int r = 0; r < 8; ++r) dacc[r] += fmaxf(c[r] + bn, 0.f) * wn;
  }
  // Reduce over the 16 lanes of each half-wave (column dimension of C).
#pragma unroll
  for (int mask = 1; mask <= 8; mask <<= 1) {
#pragma unroll
    for (int r = 0; r < 8; ++r) dacc[r] += __shfl_xor(dacc[r], mask, 32);
  }
  float b2 = sc_b2[0];
  if ((lane & 15) == 0) {
    int rb = blockrow + rowbase + ((lane & 16) ? 8 : 0);
#pragma unroll
    for (int r = 0; r < 8; ++r) out[rb + r] = dacc[r] + b2;
  }
}

extern "C" void kernel_launch(void* const* d_in, const int* in_sizes, int n_in,
                              void* d_out, int out_size, void* d_ws, size_t ws_size,
                              hipStream_t stream) {
  (void)n_in; (void)out_size; (void)ws_size;
  const float* xyz     = (const float*)d_in[0];
  const float* scores  = (const float*)d_in[1];
  const float* proprio = (const float*)d_in[2];
  const float* lang    = (const float*)d_in[3];
  const float* view    = (const float*)d_in[4];
  const float* lp_w1 = (const float*)d_in[5];  const float* lp_b1 = (const float*)d_in[6];
  const float* lp_w2 = (const float*)d_in[7];  const float* lp_b2 = (const float*)d_in[8];
  const float* np_w1 = (const float*)d_in[9];  const float* np_b1 = (const float*)d_in[10];
  const float* np_w2 = (const float*)d_in[11]; const float* np_b2 = (const float*)d_in[12];
  const float* ep_w1 = (const float*)d_in[13]; const float* ep_b1 = (const float*)d_in[14];
  const float* ep_w2 = (const float*)d_in[15]; const float* ep_b2 = (const float*)d_in[16];
  const float* sc_w1 = (const float*)d_in[17]; const float* sc_b1 = (const float*)d_in[18];
  const float* sc_w2 = (const float*)d_in[19]; const float* sc_b2 = (const float*)d_in[20];
  float* out = (float*)d_out;

  int B = in_sizes[1] / 3;   // candidate_scores is [B, K=3]

  _Float16* w = (_Float16*)d_ws;
  size_t o = 0;
  _Float16* lpW1p = w + o; o += 512 * 64;
  _Float16* lpW2p = w + o; o += 64 * 64;
  _Float16* npW1p = w + o; o += 96 * 64;
  _Float16* npW2p = w + o; o += 64 * 64;
  _Float16* epW1p = w + o; o += 160 * 64;
  _Float16* epW2p = w + o; o += 64 * 64;
  _Float16* scW1p = w + o; o += 128 * 64;
  _Float16* langH   = w + o; o += (size_t)B * 64;
  _Float16* nodeH   = w + o; o += (size_t)B * 3 * 64;
  _Float16* pooledH = w + o; o += (size_t)B * 3 * 64;

  prep_weights<<<(69632 + 255) / 256, 256, 0, stream>>>(
      lp_w1, lp_w2, np_w1, np_w2, ep_w1, ep_w2, sc_w1,
      lpW1p, lpW2p, npW1p, npW2p, epW1p, epW2p, scW1p);
  lang_kernel<<<B / 32, 256, 0, stream>>>(lang, lp_b1, lp_b2, lpW1p, lpW2p, langH);
  node_kernel<<<(B * 3) / 32, 256, 0, stream>>>(xyz, scores, proprio, view, langH,
                                                np_b1, np_b2, npW1p, npW2p, nodeH);
  edge_kernel<<<B / 16, 256, 0, stream>>>(xyz, nodeH, ep_b1, ep_b2, epW1p, epW2p, pooledH);
  score_kernel<<<(B * 3) / 64, 128, 0, stream>>>(nodeH, pooledH, sc_b1, sc_w2, sc_b2,
                                                 scW1p, out);
}
